// MetricalGNN_64338610094391
// MI455X (gfx1250) — matrix operand order repaired and checked by reading, not compiled, not run on hardware
//
#include <hip/hip_runtime.h>

typedef __attribute__((ext_vector_type(16))) _Float16 v16h;
typedef __attribute__((ext_vector_type(8)))  float    v8f;
typedef __attribute__((ext_vector_type(4)))  unsigned u32x4;
typedef __attribute__((ext_vector_type(8)))  int      i32x8;
typedef __attribute__((ext_vector_type(4)))  int      i32x4;

#define G_NN  100000
#define G_NB  20000
#define G_IN  64
#define G_HID 128
#define G_OUT 32

#define FLAG_RELU 1
#define FLAG_ACC  2

#if defined(__has_builtin)
#if __has_builtin(__builtin_amdgcn_tensor_load_to_lds)
#define USE_TDM 1
#endif
#endif

#ifdef USE_TDM
// Issue one Tensor-Data-Mover 2D tile load: rows x K f32, row stride lda,
// global -> LDS (row-major).  OOB rows (y >= rowsRemain) are zero-filled.
// D# layout per CDNA5 ISA (group0 + group1, 2D).
// This toolchain exposes the 6-arg builtin:
//   (u32x4 g0, i32x8 g1, i32x4 g2, i32x4 g3, i32x8 extra, i32 cpol)
static __device__ __forceinline__ void tdm_load_tile_f32(
    const float* gbase, unsigned ldsOff, int rowsRemain, int lda, int tileK) {
  const unsigned long long ga = (unsigned long long)(uintptr_t)gbase;
  u32x4 g0;
  g0[0] = 1u;                                         // count=1, user desc
  g0[1] = ldsOff;                                     // lds_addr (bytes)
  g0[2] = (unsigned)(ga & 0xffffffffu);               // global_addr[31:0]
  g0[3] = (unsigned)((ga >> 32) & 0x01ffffffu)        // global_addr[56:32]
          | 0x80000000u;                              // type=2 ("image")
  const unsigned d0 = (unsigned)lda;                  // tensor_dim0
  const unsigned d1 = (unsigned)rowsRemain;           // tensor_dim1 (OOB->0)
  i32x8 g1;
  g1[0] = (int)(2u << 16);                            // data_size=4B, mask=0
  g1[1] = (int)((d0 & 0xffffu) << 16);                // dim0[15:0]
  g1[2] = (int)((d0 >> 16) | ((d1 & 0xffffu) << 16)); // dim0[31:16], dim1[15:0]
  g1[3] = (int)((d1 >> 16) | ((unsigned)tileK << 16));// dim1[31:16], tile_dim0
  g1[4] = 128;                                        // tile_dim1=128, tile_dim2=0
  g1[5] = (int)d0;                                    // tensor_dim0_stride[31:0]
  g1[6] = 0;                                          // stride hi, dim1_stride lo
  g1[7] = 0;
  const i32x4 z4 = {0, 0, 0, 0};
  const i32x8 z8 = {0, 0, 0, 0, 0, 0, 0, 0};
  __builtin_amdgcn_tensor_load_to_lds(g0, g1, z4, z4, z8, 0);
}
#endif

// =====================================================================
// GEMM: C[N x M](ldc) = act( A[N x K](lda) @ W[K x M] + bias [+ C] )
// K is a template parameter (64 or 128); M % 16 == 0.
// Block = 256 threads (8 waves).  Block tile: 128 rows x 64 cols.
// Each wave computes a 16x64 strip: one A fragment feeds 4x
// v_wmma_f32_16x16x32_f16 per 32-K step (K loop fully unrolled).
// W column-block staged once into LDS pre-swizzled (B frag = 2x b128).
// A strip staged by the TDM (async DMA, overlapped with B staging) when
// available, else read directly from global with clamped rows.
// =====================================================================
template <int K>
__global__ __launch_bounds__(256)
void gemm_wmma_f16(const float* __restrict__ A, int lda,
                   const float* __restrict__ W,
                   const float* __restrict__ bias,
                   float* __restrict__ C, int ldc,
                   int N, int M, int flags)
{
  // [K/32 ksteps][4 tiles][32 owner-lanes][16 halfs]
  __shared__ __align__(32) _Float16 ldsB[K * 64];
#ifdef USE_TDM
  __shared__ __align__(16) float ldsA[128 * K];       // row-major A strip
#endif

  const int tid     = threadIdx.x;
  const int lane    = tid & 31;
  const int wave    = tid >> 5;
  const int colBase = blockIdx.x * 64;
  const int rowBase = blockIdx.y * 128 + wave * 16;
  const int kg      = lane >> 4;     // lane group (0: lanes 0-15, 1: 16-31)

#ifdef USE_TDM
  if (wave == 0) {
    const float* gA = A + (size_t)blockIdx.y * 128 * lda;
    tdm_load_tile_f32(gA, (unsigned)(uintptr_t)ldsA,
                      N - blockIdx.y * 128, lda, K);
  }
#endif

  // ---- stage W[0..K)[colBase..colBase+63] -> LDS f16 (swizzled) ----
  // Runs on all 8 waves, overlapping the TDM transfer above.
#pragma unroll
  for (int it = 0; it < (K >> 4); ++it) {
    const int e4 = tid + (it << 8);
    const int k  = e4 >> 4;                    // 0..K-1
    const int n  = (e4 & 15) << 2;             // 0..60, step 4
    const int col = colBase + n;
    float4 v = {0.f, 0.f, 0.f, 0.f};
    if (col < M) v = *(const float4*)(W + (size_t)k * M + col);
    const int ks = k >> 5;
    const int kl = k & 31;
    const int h  = (kl & 7) | ((kl >> 4) << 3);
    const int kb = ((kl >> 3) & 1) << 4;
    const int t  = n >> 4;
    const int nn = n & 15;
    _Float16* base = ldsB + (size_t)(((ks * 4 + t) * 32) + nn + kb) * 16 + h;
    base[0 * 16] = (_Float16)v.x;
    base[1 * 16] = (_Float16)v.y;
    base[2 * 16] = (_Float16)v.z;
    base[3 * 16] = (_Float16)v.w;
  }
#ifdef USE_TDM
  if (wave == 0) {
#if __has_builtin(__builtin_amdgcn_s_wait_tensorcnt)
    __builtin_amdgcn_s_wait_tensorcnt(0);
#else
    asm volatile("s_wait_tensorcnt 0x0" ::: "memory");
#endif
  }
#endif
  __syncthreads();

#ifdef USE_TDM
  const float* arow = ldsA + (size_t)(wave * 16 + (lane & 15)) * K + kg * 8;
#else
  const int mrow = rowBase + (lane & 15);
  const int mc   = (mrow < N) ? mrow : 0;      // clamp; epilogue guards stores
  const float* arow = A + (size_t)mc * lda + kg * 8;
#endif

  v8f acc[4] = {v8f{}, v8f{}, v8f{}, v8f{}};

#pragma unroll
  for (int ks = 0; ks < (K >> 5); ++ks) {
    const int kk = ks << 5;
    // ---- A fragment: 16x32 f16 layout = two contiguous 8-float runs ----
    const float4 r0 = *(const float4*)(arow + kk + 0);
    const float4 r1 = *(const float4*)(arow + kk + 4);
    const float4 r2 = *(const float4*)(arow + kk + 16);
    const float4 r3 = *(const float4*)(arow + kk + 20);
    v16h a;
    a[0]=(_Float16)r0.x;  a[1]=(_Float16)r0.y;  a[2]=(_Float16)r0.z;  a[3]=(_Float16)r0.w;
    a[4]=(_Float16)r1.x;  a[5]=(_Float16)r1.y;  a[6]=(_Float16)r1.z;  a[7]=(_Float16)r1.w;
    a[8]=(_Float16)r2.x;  a[9]=(_Float16)r2.y;  a[10]=(_Float16)r2.z; a[11]=(_Float16)r2.w;
    a[12]=(_Float16)r3.x; a[13]=(_Float16)r3.y; a[14]=(_Float16)r3.z; a[15]=(_Float16)r3.w;

    // ---- all 4 B fragments first, then 4 WMMAs (partial dscnt waits) ----
    const _Float16* bb = ldsB + (size_t)ks * 4 * 512;
    const v16h b0 = *reinterpret_cast<const v16h*>(bb + (0 * 32 + lane) * 16);
    const v16h b1 = *reinterpret_cast<const v16h*>(bb + (1 * 32 + lane) * 16);
    const v16h b2 = *reinterpret_cast<const v16h*>(bb + (2 * 32 + lane) * 16);
    const v16h b3 = *reinterpret_cast<const v16h*>(bb + (3 * 32 + lane) * 16);
    acc[0] = __builtin_amdgcn_wmma_f32_16x16x32_f16(false, a, false, b0,
                                                    (short)0, acc[0], false, false);
    acc[1] = __builtin_amdgcn_wmma_f32_16x16x32_f16(false, a, false, b1,
                                                    (short)0, acc[1], false, false);
    acc[2] = __builtin_amdgcn_wmma_f32_16x16x32_f16(false, a, false, b2,
                                                    (short)0, acc[2], false, false);
    acc[3] = __builtin_amdgcn_wmma_f32_16x16x32_f16(false, a, false, b3,
                                                    (short)0, acc[3], false, false);
  }

  // ---- epilogue: 16x16 f32 C/D layout, 4 tiles ----
#pragma unroll
  for (int t = 0; t < 4; ++t) {
    const int n = colBase + t * 16 + (lane & 15);
    if (n >= M) continue;
#pragma unroll
    for (int r = 0; r < 8; ++r) {
      const int row = rowBase + (lane >> 4) * 8 + r;
      if (row < N) {
        float v = acc[t][r];
        if (bias) v += bias[n];
        const size_t idx = (size_t)row * ldc + n;
        if (flags & FLAG_ACC)  v += C[idx];
        if (flags & FLAG_RELU) v = fmaxf(v, 0.0f);
        C[idx] = v;
      }
    }
  }
}

// =====================================================================
// Elementwise / graph kernels
// =====================================================================
__global__ void fill_kernel(float* __restrict__ p, long long n, float v) {
  long long i = (long long)blockIdx.x * blockDim.x + threadIdx.x;
  if (i < n) p[i] = v;
}

__global__ void add_inplace(float* __restrict__ y, const float* __restrict__ x,
                            long long n) {
  long long i = (long long)blockIdx.x * blockDim.x + threadIdx.x;
  if (i < n) y[i] += x[i];
}

__global__ void scale_inplace(float* __restrict__ y, float s, long long n) {
  long long i = (long long)blockIdx.x * blockDim.x + threadIdx.x;
  if (i < n) y[i] *= s;
}

// agg[dst] += X[src] over edges; cnt[dst] += 1.  One thread = (edge, 4 ch).
__global__ void scatter_add(const int* __restrict__ src,
                            const int* __restrict__ dst,
                            const float* __restrict__ X, int ldx,
                            float* __restrict__ agg, float* __restrict__ cnt,
                            int E, int K)
{
  const int nblk = K >> 2;
  const int tid = blockIdx.x * blockDim.x + threadIdx.x;
  if (tid >= E * nblk) return;
  const int e = tid / nblk;
  const int j = (tid - e * nblk) << 2;
  const int s = src[e];
  const int d = dst[e];
  float* out = agg + (size_t)d * K + j;
  __builtin_prefetch(out, 1, 1);                       // global_prefetch_b8
  const float4 v = *(const float4*)(X + (size_t)s * ldx + j);
  atomicAdd(out + 0, v.x);
  atomicAdd(out + 1, v.y);
  atomicAdd(out + 2, v.z);
  atomicAdd(out + 3, v.w);
  if (j == 0) atomicAdd(cnt + d, 1.0f);
}

__global__ void div_cnt(float* __restrict__ agg, const float* __restrict__ cnt,
                        int N, int K) {
  const int tid = blockIdx.x * blockDim.x + threadIdx.x;
  if (tid >= N * K) return;
  const int n = tid / K;
  agg[tid] /= fmaxf(cnt[n], 1.0f);
}

__device__ __forceinline__ float waveRedSum(float v) {
#pragma unroll
  for (int off = 16; off; off >>= 1) v += __shfl_xor(v, off, 32);
  return v;
}

// Row L2-normalize, M = 128 (one wave per row, 4 floats per lane)
__global__ __launch_bounds__(256)
void l2norm_rows(float* __restrict__ X, int N) {
  const int row  = blockIdx.x * 8 + (threadIdx.x >> 5);
  const int lane = threadIdx.x & 31;
  if (row >= N) return;
  float4 v = *(float4*)(X + (size_t)row * G_HID + lane * 4);
  float ss = v.x * v.x + v.y * v.y + v.z * v.z + v.w * v.w;
  ss = waveRedSum(ss);
  const float inv = 1.0f / fmaxf(sqrtf(ss), 1e-12f);
  v.x *= inv; v.y *= inv; v.z *= inv; v.w *= inv;
  *(float4*)(X + (size_t)row * G_HID + lane * 4) = v;
}

// x = LN(relu(x)) * g + b over rows of width 128
__global__ __launch_bounds__(256)
void relu_ln_rows(float* __restrict__ X, const float* __restrict__ g,
                  const float* __restrict__ b, int N) {
  const int row  = blockIdx.x * 8 + (threadIdx.x >> 5);
  const int lane = threadIdx.x & 31;
  if (row >= N) return;
  float4 v = *(float4*)(X + (size_t)row * G_HID + lane * 4);
  v.x = fmaxf(v.x, 0.f); v.y = fmaxf(v.y, 0.f);
  v.z = fmaxf(v.z, 0.f); v.w = fmaxf(v.w, 0.f);
  float s  = v.x + v.y + v.z + v.w;
  float s2 = v.x * v.x + v.y * v.y + v.z * v.z + v.w * v.w;
  s  = waveRedSum(s);
  s2 = waveRedSum(s2);
  const float mean = s * (1.0f / G_HID);
  const float var  = s2 * (1.0f / G_HID) - mean * mean;
  const float inv  = rsqrtf(var + 1e-5f);
  const int c = lane * 4;
  v.x = (v.x - mean) * inv * g[c + 0] + b[c + 0];
  v.y = (v.y - mean) * inv * g[c + 1] + b[c + 1];
  v.z = (v.z - mean) * inv * g[c + 2] + b[c + 2];
  v.w = (v.w - mean) * inv * g[c + 3] + b[c + 3];
  *(float4*)(X + (size_t)row * G_HID + lane * 4) = v;
}

// BatchNorm eval (running mean=0, var=1): x = x * g/sqrt(1+eps) + b, width 128
__global__ void bn_rows(float* __restrict__ X, const float* __restrict__ g,
                        const float* __restrict__ b, long long n) {
  long long i = (long long)blockIdx.x * blockDim.x + threadIdx.x;
  if (i >= n) return;
  const int c = (int)(i & (G_HID - 1));
  X[i] = X[i] * (g[c] * rsqrtf(1.0f + 1e-5f)) + b[c];
}

// =====================================================================
// Host orchestration
// =====================================================================
extern "C" void kernel_launch(void* const* d_in, const int* in_sizes, int n_in,
                              void* d_out, int out_size, void* d_ws, size_t ws_size,
                              hipStream_t stream) {
  (void)in_sizes; (void)n_in; (void)out_size; (void)ws_size;

  const float* x_note = (const float*)d_in[0];
  const float* x_beat = (const float*)d_in[1];
  const int*   e_ptr[5] = {(const int*)d_in[2], (const int*)d_in[3],
                           (const int*)d_in[4], (const int*)d_in[5],
                           (const int*)d_in[6]};
  const int    E[5]    = {800000, 300000, 100000, 100000, 20000};
  const int    rsrc[5] = {0, 0, 0, 1, 1};   // 0 = note, 1 = beat
  const int    rdst[5] = {0, 0, 1, 0, 1};

  const float* proj_W = (const float*)d_in[7];
  const float* proj_b = (const float*)d_in[8];
  const float* l0_Wl  = (const float*)d_in[9];
  const float* l0_bl  = (const float*)d_in[10];
  const float* l0_Wr  = (const float*)d_in[11];
  const float* Wl     = (const float*)d_in[12];
  const float* bl     = (const float*)d_in[13];
  const float* Wr     = (const float*)d_in[14];
  const float* ln_g   = (const float*)d_in[15];
  const float* ln_b   = (const float*)d_in[16];
  const float* mlp_W1 = (const float*)d_in[17];
  const float* mlp_b1 = (const float*)d_in[18];
  const float* bn_g   = (const float*)d_in[19];
  const float* bn_b   = (const float*)d_in[20];
  const float* mlp_W2 = (const float*)d_in[21];
  const float* mlp_b2 = (const float*)d_in[22];

  // workspace layout
  float* ws = (float*)d_ws;
  size_t off = 0;
  auto walloc = [&](size_t n) { float* p = ws + off; off += n; return p; };
  float* noteA = walloc((size_t)G_NN * G_HID);
  float* noteB = walloc((size_t)G_NN * G_HID);
  float* beatA = walloc((size_t)G_NB * G_HID);
  float* beatB = walloc((size_t)G_NB * G_HID);
  float* tmp   = walloc((size_t)G_NN * G_HID);   // proj src / per-relation output
  float* agg   = walloc((size_t)G_NN * G_HID);   // segment mean / mlp hidden
  float* cnt   = walloc((size_t)G_NN);

  auto fill = [&](float* p, long long n, float v) {
    fill_kernel<<<dim3((unsigned)((n + 255) / 256)), 256, 0, stream>>>(p, n, v);
  };
  auto gemm = [&](const float* A, int lda, const float* W_, const float* bias,
                  float* C, int ldc, int N, int K, int M, int flags) {
    dim3 grid((M + 63) / 64, (N + 127) / 128);
    if (K == 64)
      gemm_wmma_f16<64><<<grid, 256, 0, stream>>>(A, lda, W_, bias, C, ldc,
                                                  N, M, flags);
    else
      gemm_wmma_f16<128><<<grid, 256, 0, stream>>>(A, lda, W_, bias, C, ldc,
                                                   N, M, flags);
  };

  const float* cur[2] = {x_note, x_beat};
  float* nxt[2] = {noteA, beatA};
  float* alt[2] = {noteB, beatB};
  const int nNodes[2] = {G_NN, G_NB};

  for (int i = 0; i < 3; ++i) {
    const int Kin = (i == 0) ? G_IN : G_HID;
    fill(nxt[0], (long long)G_NN * G_HID, 0.f);
    fill(nxt[1], (long long)G_NB * G_HID, 0.f);

    for (int r = 0; r < 5; ++r) {
      const int s = rsrc[r], d = rdst[r];
      const int Ns = nNodes[s], Nd = nNodes[d];
      const float* xs = cur[s];
      const float* xd = cur[d];

      const float* gsrc;
      if (i == 0) {  // SAGEConv project=True: xs' = relu(xs @ proj_W[r] + b)
        gemm(xs, Kin, proj_W + (size_t)r * 64 * 64, proj_b + (size_t)r * 64,
             tmp, 64, Ns, 64, 64, FLAG_RELU);
        gsrc = tmp;
      } else {
        gsrc = xs;
      }

      // segment mean: agg[dst] = mean_{edges} gsrc[src]
      fill(agg, (long long)Nd * Kin, 0.f);
      fill(cnt, Nd, 0.f);
      const int tot = E[r] * (Kin >> 2);
      scatter_add<<<dim3((tot + 255) / 256), 256, 0, stream>>>(
          e_ptr[r], e_ptr[r] + E[r], gsrc, Kin, agg, cnt, E[r], Kin);
      div_cnt<<<dim3((Nd * Kin + 255) / 256), 256, 0, stream>>>(agg, cnt, Nd, Kin);

      const float *Wl_, *bl_, *Wr_;
      if (i == 0) {
        Wl_ = l0_Wl + (size_t)r * 64 * 128;
        bl_ = l0_bl + (size_t)r * 128;
        Wr_ = l0_Wr + (size_t)r * 64 * 128;
      } else {
        const size_t o5 = (size_t)(i - 1) * 5 + r;
        Wl_ = Wl + o5 * 128 * 128;
        bl_ = bl + o5 * 128;
        Wr_ = Wr + o5 * 128 * 128;
      }

      // o = agg @ W_l + b_l + xd @ W_r      (tmp is free again after scatter)
      gemm(agg, Kin, Wl_, bl_, tmp, G_HID, Nd, Kin, G_HID, 0);
      gemm(xd,  Kin, Wr_, nullptr, tmp, G_HID, Nd, Kin, G_HID, FLAG_ACC);
      if (i == 0)
        l2norm_rows<<<dim3((Nd + 7) / 8), 256, 0, stream>>>(tmp, Nd);

      add_inplace<<<dim3((unsigned)(((long long)Nd * G_HID + 255) / 256)), 256, 0,
                    stream>>>(nxt[d], tmp, (long long)Nd * G_HID);
    }

    // HeteroConv mean over relations: note gets 3 rels, beat gets 2
    scale_inplace<<<dim3((unsigned)(((long long)G_NN * G_HID + 255) / 256)), 256, 0,
                    stream>>>(nxt[0], 1.0f / 3.0f, (long long)G_NN * G_HID);
    scale_inplace<<<dim3((unsigned)(((long long)G_NB * G_HID + 255) / 256)), 256, 0,
                    stream>>>(nxt[1], 0.5f, (long long)G_NB * G_HID);

    if (i != 2) {
      relu_ln_rows<<<dim3((G_NN + 7) / 8), 256, 0, stream>>>(
          nxt[0], ln_g + (size_t)i * G_HID, ln_b + (size_t)i * G_HID, G_NN);
      relu_ln_rows<<<dim3((G_NB + 7) / 8), 256, 0, stream>>>(
          nxt[1], ln_g + (size_t)i * G_HID, ln_b + (size_t)i * G_HID, G_NB);
    }

    cur[0] = nxt[0]; cur[1] = nxt[1];
    float* t0 = nxt[0]; float* t1 = nxt[1];
    nxt[0] = alt[0]; nxt[1] = alt[1];
    alt[0] = t0;     alt[1] = t1;
  }

  // MLP head on note nodes: h = relu(x @ W1 + b1); BN(eval); out = h @ W2 + b2
  float* hid = agg;  // agg is free now
  gemm(cur[0], G_HID, mlp_W1, mlp_b1, hid, G_HID, G_NN, G_HID, G_HID, FLAG_RELU);
  bn_rows<<<dim3((unsigned)(((long long)G_NN * G_HID + 255) / 256)), 256, 0,
            stream>>>(hid, bn_g, bn_b, (long long)G_NN * G_HID);
  gemm(hid, G_HID, mlp_W2, mlp_b2, (float*)d_out, G_OUT, G_NN, G_HID, G_OUT, 0);
}
